// Bond2Bond_graph_Transformer_12730283065996
// MI455X (gfx1250) — matrix operand order repaired and checked by reading, compiled, not run on hardware
//
#include <hip/hip_runtime.h>
#include <hip/hip_bf16.h>
#include <math.h>

#define NN 40000
#define EE 640000
#define DD 256
#define HH 8
#define DHH 32
#define DFF 1024
#define SCALE 0.0625f   // 256^-0.5
#define EPSV 1e-5f

typedef __attribute__((ext_vector_type(16))) __bf16 v16bf;
typedef __attribute__((ext_vector_type(8)))  float  v8f;

// ---------- helpers ----------
static __device__ inline unsigned short f2bf(float f) {
    unsigned u = __float_as_uint(f);
    u += 0x7FFFu + ((u >> 16) & 1u);          // round-to-nearest-even
    return (unsigned short)(u >> 16);
}
static __device__ inline float bflo(unsigned p) { return __uint_as_float(p << 16); }
static __device__ inline float bfhi(unsigned p) { return __uint_as_float(p & 0xFFFF0000u); }

static __device__ inline void atomicMaxF32(float* addr, float v) {
    if (v >= 0.0f) atomicMax((int*)addr, __float_as_int(v));
    else           atomicMin((unsigned int*)addr, __float_as_uint(v));
}

// One 16x32 C macro-tile per wave (two 16x16 WMMA tiles sharing the A fragment).
// A: row-major bf16 (stride K). Bt: column-major bf16 weights (Bt[n*K + k]).
// A and B fragments both use contiguous k order (k = 16*half + s); C uses row 8*half + r.
template <int K>
static __device__ inline void gemm_tile2_bf16(const unsigned short* __restrict__ A,
                                              const unsigned short* __restrict__ Bt,
                                              int row0, int col0, int lane,
                                              v8f& acc0, v8f& acc1) {
    const int m    = lane & 15;
    const int half = lane >> 4;
    const unsigned* __restrict__ arow  = (const unsigned*)(A  + (size_t)(row0 + m) * K);
    const unsigned* __restrict__ brow0 = (const unsigned*)(Bt + (size_t)(col0 + m) * K);
    const unsigned* __restrict__ brow1 = (const unsigned*)(Bt + (size_t)(col0 + 16 + m) * K);
    const int akb = half * 8;   // A: 16 bf16 offset (in u32 units)
    const int bkb = half * 8;   // B: 16 bf16 offset (in u32 units)
#pragma unroll
    for (int k0 = 0; k0 < K; k0 += 32) {
        const int base = k0 >> 1;                       // u32 index
        union { unsigned u[8]; v16bf v; } a, b0, b1;
#pragma unroll
        for (int g = 0; g < 4; ++g) a.u[g]     = arow[base + akb + g];      // K: k0+0..7 / +16..23
#pragma unroll
        for (int g = 0; g < 4; ++g) a.u[4 + g] = arow[base + 4 + akb + g];  // K: k0+8..15 / +24..31
#pragma unroll
        for (int g = 0; g < 8; ++g) b0.u[g]    = brow0[base + bkb + g];     // K: k0+0..15 / +16..31
#pragma unroll
        for (int g = 0; g < 8; ++g) b1.u[g]    = brow1[base + bkb + g];
        if (k0 + 32 < K) {                              // WGP-scope global_prefetch_b8
            __builtin_prefetch(arow  + base + 16, 0, 3);
            __builtin_prefetch(brow0 + base + 16, 0, 3);
            __builtin_prefetch(brow1 + base + 16, 0, 3);
        }
        acc0 = __builtin_amdgcn_wmma_f32_16x16x32_bf16(false, a.v, false, b0.v,
                                                       (short)0, acc0, false, false);
        acc1 = __builtin_amdgcn_wmma_f32_16x16x32_bf16(false, a.v, false, b1.v,
                                                       (short)0, acc1, false, false);
    }
}

// ---------- kernel 0: convert + transpose weights to bf16 column-major ----------
__global__ void k_prep_weights(const float* __restrict__ Wqkv, const float* __restrict__ W_in,
                               const float* __restrict__ W_out,
                               unsigned short* __restrict__ wqkvT,
                               unsigned short* __restrict__ winT,
                               unsigned short* __restrict__ woutT) {
    int i = blockIdx.x * blockDim.x + threadIdx.x;
    const int NQ = 768 * 256, NI = 1024 * 256, NO = 256 * 1024;
    if (i < NQ) { int n = i / 256, k = i % 256;  wqkvT[i] = f2bf(Wqkv[k * 768 + n]);  return; }
    i -= NQ;
    if (i < NI) { int n = i / 256, k = i % 256;  winT[i]  = f2bf(W_in[k * 1024 + n]); return; }
    i -= NI;
    if (i < NO) { int n = i / 1024, k = i % 1024; woutT[i] = f2bf(W_out[k * 256 + n]); }
}

// ---------- kernel 1: init segment-max / segment-sum / agg ----------
__global__ void k_init(float* __restrict__ m, float* __restrict__ s, float* __restrict__ agg) {
    int i = blockIdx.x * blockDim.x + threadIdx.x;
    if (i < NN * HH) { m[i] = -INFINITY; s[i] = 0.0f; }
    if (i < NN * DD) agg[i] = 0.0f;
}

// ---------- wave-per-row LayerNorm -> bf16 ----------
__global__ void k_ln_attn(const float* __restrict__ x, const float* __restrict__ g,
                          const float* __restrict__ b, unsigned short* __restrict__ out) {
    const int lane = threadIdx.x & 31;
    const int row  = blockIdx.x * 8 + (threadIdx.x >> 5);
    const float4* p = (const float4*)(x + (size_t)row * DD + lane * 8);
    float4 v0 = p[0], v1 = p[1];
    float sum = v0.x + v0.y + v0.z + v0.w + v1.x + v1.y + v1.z + v1.w;
    float sq  = v0.x*v0.x + v0.y*v0.y + v0.z*v0.z + v0.w*v0.w
              + v1.x*v1.x + v1.y*v1.y + v1.z*v1.z + v1.w*v1.w;
#pragma unroll
    for (int off = 16; off > 0; off >>= 1) {
        sum += __shfl_xor(sum, off, 32);
        sq  += __shfl_xor(sq,  off, 32);
    }
    const float mu = sum * (1.0f / DD);
    const float rstd = rsqrtf(sq * (1.0f / DD) - mu * mu + EPSV);
    const int c = lane * 8;
    float y[8] = { v0.x, v0.y, v0.z, v0.w, v1.x, v1.y, v1.z, v1.w };
    unsigned pk[4];
#pragma unroll
    for (int j = 0; j < 4; ++j) {
        float a0 = (y[2*j]   - mu) * rstd * g[c + 2*j]     + b[c + 2*j];
        float a1 = (y[2*j+1] - mu) * rstd * g[c + 2*j + 1] + b[c + 2*j + 1];
        pk[j] = (unsigned)f2bf(a0) | ((unsigned)f2bf(a1) << 16);
    }
    unsigned* o = (unsigned*)(out + (size_t)row * DD + c);
#pragma unroll
    for (int j = 0; j < 4; ++j) o[j] = pk[j];
}

// ---------- residual add + LayerNorm ----------
__global__ void k_res_ln(const float* __restrict__ h0, const float* __restrict__ agg,
                         const float* __restrict__ g, const float* __restrict__ b,
                         float* __restrict__ xout, unsigned short* __restrict__ xln) {
    const int lane = threadIdx.x & 31;
    const int row  = blockIdx.x * 8 + (threadIdx.x >> 5);
    const size_t base = (size_t)row * DD + lane * 8;
    float y[8];
    float sum = 0.0f, sq = 0.0f;
#pragma unroll
    for (int j = 0; j < 8; ++j) {
        float v = h0[base + j] + agg[base + j];
        y[j] = v; sum += v; sq += v * v;
        xout[base + j] = v;
    }
#pragma unroll
    for (int off = 16; off > 0; off >>= 1) {
        sum += __shfl_xor(sum, off, 32);
        sq  += __shfl_xor(sq,  off, 32);
    }
    const float mu = sum * (1.0f / DD);
    const float rstd = rsqrtf(sq * (1.0f / DD) - mu * mu + EPSV);
    const int c = lane * 8;
    unsigned* o = (unsigned*)(xln + (size_t)row * DD + c);
#pragma unroll
    for (int j = 0; j < 4; ++j) {
        float a0 = (y[2*j]   - mu) * rstd * g[c + 2*j]     + b[c + 2*j];
        float a1 = (y[2*j+1] - mu) * rstd * g[c + 2*j + 1] + b[c + 2*j + 1];
        o[j] = (unsigned)f2bf(a0) | ((unsigned)f2bf(a1) << 16);
    }
}

// ---------- QKV projection (WMMA): qkv = LN(h) @ Wqkv + bqkv ; q *= SCALE ----------
__global__ void k_qkv(const unsigned short* __restrict__ hbf,
                      const unsigned short* __restrict__ wqkvT,
                      const float* __restrict__ bqkv,
                      unsigned short* __restrict__ qkv) {
    const int lane = threadIdx.x & 31;
    const int wave = threadIdx.x >> 5;
    const int row0 = blockIdx.x * 16;
    const int col0 = (blockIdx.y * 8 + wave) * 32;
    v8f acc0 = {}, acc1 = {};
    gemm_tile2_bf16<256>(hbf, wqkvT, row0, col0, lane, acc0, acc1);
    const int half = lane >> 4;
#pragma unroll
    for (int t = 0; t < 2; ++t) {
        const int col = col0 + t * 16 + (lane & 15);
        const float bias = bqkv[col];
        const float scl = (col < 256) ? SCALE : 1.0f;
        const v8f& acc = t ? acc1 : acc0;
#pragma unroll
        for (int r = 0; r < 8; ++r) {
            int row = row0 + half * 8 + r;
            qkv[(size_t)row * 768 + col] = f2bf((acc[r] + bias) * scl);
        }
    }
}

// ---------- edge scores + segment max (wave per edge) ----------
__global__ void k_scores(const unsigned short* __restrict__ qkv,
                         const int* __restrict__ src, const int* __restrict__ dst,
                         float* __restrict__ scores, float* __restrict__ mbuf) {
    const int lane = threadIdx.x & 31;
    const int e = blockIdx.x * 8 + (threadIdx.x >> 5);
    const int h = lane >> 2, sub = lane & 3;
    const int se = src[e], de = dst[e];
    const unsigned* qp = (const unsigned*)(qkv + (size_t)se * 768 + h * DHH + sub * 8);
    const unsigned* kp = (const unsigned*)(qkv + (size_t)de * 768 + 256 + h * DHH + sub * 8);
    float acc = 0.0f;
#pragma unroll
    for (int j = 0; j < 4; ++j) {
        unsigned qa = qp[j], ka = kp[j];
        acc += bflo(qa) * bflo(ka) + bfhi(qa) * bfhi(ka);
    }
    acc += __shfl_xor(acc, 1, 32);
    acc += __shfl_xor(acc, 2, 32);
    if (sub == 0) {
        scores[(size_t)e * HH + h] = acc;
        atomicMaxF32(&mbuf[(size_t)de * HH + h], acc);
    }
}

// ---------- exp(score - max) + segment sum (in-place on scores) ----------
__global__ void k_softnorm(const int* __restrict__ dst, const float* __restrict__ mbuf,
                           float* __restrict__ scores, float* __restrict__ sbuf) {
    const int i = blockIdx.x * blockDim.x + threadIdx.x;   // i in [0, E*H)
    const int e = i >> 3, h = i & 7;
    const int d = dst[e];
    const float ex = expf(scores[i] - mbuf[(size_t)d * HH + h]);
    scores[i] = ex;
    atomicAdd(&sbuf[(size_t)d * HH + h], ex);
}

// ---------- weighted V aggregation: agg[dst] += v[src] * sa (wave per edge) ----------
__global__ void k_aggregate(const unsigned short* __restrict__ qkv,
                            const int* __restrict__ src, const int* __restrict__ dst,
                            const float* __restrict__ escore, const float* __restrict__ sbuf,
                            float* __restrict__ agg) {
    const int lane = threadIdx.x & 31;
    const int e = blockIdx.x * 8 + (threadIdx.x >> 5);
    const int h = lane >> 2, sub = lane & 3;
    const int se = src[e], de = dst[e];
    const float sa = escore[(size_t)e * HH + h] / sbuf[(size_t)de * HH + h];
    const unsigned* vp = (const unsigned*)(qkv + (size_t)se * 768 + 512 + h * DHH + sub * 8);
    float* ap = agg + (size_t)de * DD + h * DHH + sub * 8;
#pragma unroll
    for (int j = 0; j < 4; ++j) {
        unsigned va = vp[j];
        atomicAdd(ap + 2 * j,     bflo(va) * sa);
        atomicAdd(ap + 2 * j + 1, bfhi(va) * sa);
    }
}

// ---------- FFN GEMM1 (WMMA) + exact GELU ----------
__global__ void k_ffn1(const unsigned short* __restrict__ xln,
                       const unsigned short* __restrict__ winT,
                       const float* __restrict__ b_in,
                       unsigned short* __restrict__ act) {
    const int lane = threadIdx.x & 31;
    const int wave = threadIdx.x >> 5;
    const int row0 = blockIdx.x * 16;
    const int col0 = (blockIdx.y * 8 + wave) * 32;
    v8f acc0 = {}, acc1 = {};
    gemm_tile2_bf16<256>(xln, winT, row0, col0, lane, acc0, acc1);
    const int half = lane >> 4;
#pragma unroll
    for (int t = 0; t < 2; ++t) {
        const int col = col0 + t * 16 + (lane & 15);
        const float bias = b_in[col];
        const v8f& acc = t ? acc1 : acc0;
#pragma unroll
        for (int r = 0; r < 8; ++r) {
            int row = row0 + half * 8 + r;
            float v = acc[r] + bias;
            v = 0.5f * v * (1.0f + erff(v * 0.70710678118f));   // exact GELU
            act[(size_t)row * DFF + col] = f2bf(v);
        }
    }
}

// ---------- FFN GEMM2 (WMMA) + residual -> output ----------
__global__ void k_ffn2(const unsigned short* __restrict__ act,
                       const unsigned short* __restrict__ woutT,
                       const float* __restrict__ b_out, const float* __restrict__ x,
                       float* __restrict__ out) {
    const int lane = threadIdx.x & 31;
    const int wave = threadIdx.x >> 5;
    const int row0 = blockIdx.x * 16;
    const int col0 = (blockIdx.y * 8 + wave) * 32;
    v8f acc0 = {}, acc1 = {};
    gemm_tile2_bf16<1024>(act, woutT, row0, col0, lane, acc0, acc1);
    const int half = lane >> 4;
#pragma unroll
    for (int t = 0; t < 2; ++t) {
        const int col = col0 + t * 16 + (lane & 15);
        const float bias = b_out[col];
        const v8f& acc = t ? acc1 : acc0;
#pragma unroll
        for (int r = 0; r < 8; ++r) {
            int row = row0 + half * 8 + r;
            size_t idx = (size_t)row * DD + col;
            out[idx] = x[idx] + acc[r] + bias;
        }
    }
}

// ---------- host launch ----------
static inline size_t alignup(size_t v) { return (v + 255) & ~(size_t)255; }

extern "C" void kernel_launch(void* const* d_in, const int* in_sizes, int n_in,
                              void* d_out, int out_size, void* d_ws, size_t ws_size,
                              hipStream_t stream) {
    const float* triplet = (const float*)d_in[0];
    const int*   src     = (const int*)  d_in[1];
    const int*   dst     = (const int*)  d_in[2];
    const float* Wqkv    = (const float*)d_in[3];
    const float* bqkv    = (const float*)d_in[4];
    const float* g1      = (const float*)d_in[5];
    const float* b1      = (const float*)d_in[6];
    const float* g2      = (const float*)d_in[7];
    const float* b2      = (const float*)d_in[8];
    const float* W_in    = (const float*)d_in[9];
    const float* b_in    = (const float*)d_in[10];
    const float* W_out   = (const float*)d_in[11];
    const float* b_out   = (const float*)d_in[12];
    float* out = (float*)d_out;

    char* ws = (char*)d_ws;
    size_t off = 0;
    auto take = [&](size_t bytes) { char* p = ws + off; off += alignup(bytes); return p; };
    unsigned short* hbf   = (unsigned short*)take((size_t)NN * DD * 2);
    unsigned short* wqkvT = (unsigned short*)take((size_t)768 * 256 * 2);
    unsigned short* winT  = (unsigned short*)take((size_t)1024 * 256 * 2);
    unsigned short* woutT = (unsigned short*)take((size_t)256 * 1024 * 2);
    unsigned short* qkv   = (unsigned short*)take((size_t)NN * 768 * 2);
    float*          sc    = (float*)take((size_t)EE * HH * 4);   // scores, then exp in place
    float*          mbuf  = (float*)take((size_t)NN * HH * 4);
    float*          sbuf  = (float*)take((size_t)NN * HH * 4);
    float*          agg   = (float*)take((size_t)NN * DD * 4);
    float*          xbuf  = (float*)take((size_t)NN * DD * 4);
    unsigned short* xln   = (unsigned short*)take((size_t)NN * DD * 2);
    unsigned short* act   = (unsigned short*)take((size_t)NN * DFF * 2);
    (void)ws_size; (void)n_in; (void)in_sizes; (void)out_size;

    const int prep_total = 768 * 256 + 1024 * 256 + 256 * 1024;
    k_prep_weights<<<(prep_total + 255) / 256, 256, 0, stream>>>(Wqkv, W_in, W_out,
                                                                 wqkvT, winT, woutT);
    k_init<<<(NN * DD + 255) / 256, 256, 0, stream>>>(mbuf, sbuf, agg);
    k_ln_attn<<<NN / 8, 256, 0, stream>>>(triplet, g1, b1, hbf);
    k_qkv<<<dim3(NN / 16, 768 / 256), 256, 0, stream>>>(hbf, wqkvT, bqkv, qkv);
    k_scores<<<EE / 8, 256, 0, stream>>>(qkv, src, dst, sc, mbuf);
    k_softnorm<<<(EE * HH) / 256, 256, 0, stream>>>(dst, mbuf, sc, sbuf);
    k_aggregate<<<EE / 8, 256, 0, stream>>>(qkv, src, dst, sc, sbuf, agg);
    k_res_ln<<<NN / 8, 256, 0, stream>>>(triplet, agg, g2, b2, xbuf, xln);
    k_ffn1<<<dim3(NN / 16, DFF / 256), 256, 0, stream>>>(xln, winT, b_in, act);
    k_ffn2<<<dim3(NN / 16, DD / 256), 256, 0, stream>>>(act, woutT, b_out, xbuf, out);
}
